// HyperbolicTemporalAttention_47596827574591
// MI455X (gfx1250) — compile-verified, matrix operand order checked by reading, not run on hardware
//
#include <hip/hip_runtime.h>
#include <hip/hip_bf16.h>
#include <math.h>

// HyperbolicTemporalAttention for MI455X (gfx1250, wave32).
// N=32768, H=32, D=64. Streaming (online-softmax) single pass over history:
// 268 MB read once -> ~12us HBM roofline; keys GEMM via v_wmma_f32_16x16x32_f16
// (f32 accum) keeps compute well under the memory time. h-loop kept rolled,
// single tile buffer + depth-2 global_prefetch pipeline, query packed f16,
// launch_bounds(256,4): <=256 VGPRs, 4 waves/SIMD, no spills.

#define NN 32768
#define HH 32
#define DD 64

typedef __attribute__((ext_vector_type(16))) _Float16 v16h;
typedef __attribute__((ext_vector_type(8)))  float    v8f;

__device__ __forceinline__ v8f wmma_f16(v16h a, v16h b, v8f c) {
  // D = A(16x32 f16) x B(32x16 f16) + C(16x16 f32)
  return __builtin_amdgcn_wmma_f32_16x16x32_f16(
      /*neg_a=*/false, a, /*neg_b=*/false, b,
      /*c_mod=*/(short)0, c, /*reuse_a=*/false, /*reuse_b=*/false);
}

// Hardware transcendental tanh (gfx1250 V_TANH_F32). TRANS ops need one
// independent op / V_NOP before the result is consumed (ISA 7.4.3), so the
// asm fallback embeds a v_nop. The builtin (if declared) handles this itself.
__device__ __forceinline__ float htanh(float x) {
#if __has_builtin(__builtin_amdgcn_tanhf)
  return __builtin_amdgcn_tanhf(x);
#else
  float y;
  asm("v_tanh_f32 %0, %1\n\tv_nop" : "=v"(y) : "v"(x));
  return y;
#endif
}

// Lane l owns row (l&15); kb=(l>>4)*8 selects which K-interleave half it holds.
// Chunk c (c=0..3) = 8 contiguous floats at column c*16+kb. Chunks 0,1 are
// exactly A-frag elements 0..15 for K in [0,32); chunks 2,3 for K in [32,64).
__device__ __forceinline__ void load_tile(float t[32], const float* rowp) {
#pragma unroll
  for (int c2 = 0; c2 < 4; ++c2) {
    float4 x = *(const float4*)(rowp + c2 * 16);
    float4 y = *(const float4*)(rowp + c2 * 16 + 4);
    t[c2 * 8 + 0] = x.x; t[c2 * 8 + 1] = x.y; t[c2 * 8 + 2] = x.z; t[c2 * 8 + 3] = x.w;
    t[c2 * 8 + 4] = y.x; t[c2 * 8 + 5] = y.y; t[c2 * 8 + 6] = y.z; t[c2 * 8 + 7] = y.w;
  }
}

// B fragment of W^T (keys[n,e] = sum_d T[n,d]*W[e,d]): B[k=d][col=e], col = t4*16 + (l&15).
// Element j<8 -> K=kc*32+kb+j ; j>=8 -> K=kc*32+kb+16+(j-8)  (mirrors A layout).
__device__ __forceinline__ void load_wfrags(const float* W, int lq, int kb, v16h frag[4][2]) {
#pragma unroll
  for (int t4 = 0; t4 < 4; ++t4) {
    const float* wrow = W + (t4 * 16 + lq) * DD;
#pragma unroll
    for (int kc = 0; kc < 2; ++kc) {
      const float* p = wrow + kc * 32 + kb;
      float4 x0 = *(const float4*)(p);
      float4 x1 = *(const float4*)(p + 4);
      float4 y0 = *(const float4*)(p + 16);
      float4 y1 = *(const float4*)(p + 20);
      float tmp[16] = {x0.x, x0.y, x0.z, x0.w, x1.x, x1.y, x1.z, x1.w,
                       y0.x, y0.y, y0.z, y0.w, y1.x, y1.y, y1.z, y1.w};
      v16h f;
#pragma unroll
      for (int j = 0; j < 16; ++j) f[j] = (_Float16)tmp[j];
      frag[t4][kc] = f;
    }
  }
}

__device__ __forceinline__ void process_tile(
    float t[32], int parity, int lq, int half,
    const v16h Bk[4][2], const v16h qh0, const v16h qh1,
    const float bkT[4], const float avT[4],
    float sqrt_c, float tos,
    float mrun[8], float lrun[8], float acc[32],
    volatile float* swb) {
  // ---- log map at origin: per-row radial scale (row split across lane pair l, l^16)
  float ss = 0.f;
#pragma unroll
  for (int j = 0; j < 32; ++j) ss += t[j] * t[j];
  ss += __shfl_xor(ss, 16, 32);
  float nrm = sqrtf(ss);
  float sc = tos * atanhf(sqrt_c * nrm) / nrm;
#pragma unroll
  for (int j = 0; j < 32; ++j) t[j] *= sc;

  // ---- A fragments (f16) straight from chunk ordering
  v16h a0, a1;
#pragma unroll
  for (int j = 0; j < 16; ++j) { a0[j] = (_Float16)t[j]; a1[j] = (_Float16)t[16 + j]; }

  // ---- keys = T @ Wk^T + bk (WMMA, f32 accum), logits partial
  float p[8];
#pragma unroll
  for (int r = 0; r < 8; ++r) p[r] = 0.f;
#pragma unroll
  for (int t4 = 0; t4 < 4; ++t4) {
    v8f kf;
#pragma unroll
    for (int r = 0; r < 8; ++r) kf[r] = bkT[t4];  // bias: C element (m,e) = bk[e]
    kf = wmma_f16(a0, Bk[t4][0], kf);
    kf = wmma_f16(a1, Bk[t4][1], kf);
#pragma unroll
    for (int r = 0; r < 8; ++r) {
      float qv = (float)((t4 < 2) ? qh0[(t4 & 1) * 8 + r] : qh1[(t4 & 1) * 8 + r]);
      p[r] += htanh(qv + kf[r]) * avT[t4];
    }
  }
  // reduce over the 16 column-lanes of each half (rows m = r + 8*half)
#pragma unroll
  for (int r = 0; r < 8; ++r) {
    float v = p[r];
    v += __shfl_xor(v, 1, 32);
    v += __shfl_xor(v, 2, 32);
    v += __shfl_xor(v, 4, 32);
    v += __shfl_xor(v, 8, 32);
    p[r] = v;
  }

  // ---- online softmax over h
  float wv8[8], cv8[8];
#pragma unroll
  for (int r = 0; r < 8; ++r) {
    float mo = mrun[r];
    float mn = fmaxf(mo, p[r]);
    float corr = __expf(mo - mn);
    float w = __expf(p[r] - mn);
    lrun[r] = lrun[r] * corr + w;
    mrun[r] = mn;
    wv8[r] = w;
    cv8[r] = corr;
  }

  // ---- broadcast per-row w/corr from D-frag ownership to A-frag ownership.
  // Per-wave private LDS slice; DS ops are in-order within a wave, and the
  // store instruction precedes the loads in the single wave32 instruction
  // stream, so no workgroup barrier is required.
  volatile float* base = swb + parity * 32;
  if (lq == 0) {
#pragma unroll
    for (int r = 0; r < 8; ++r) {
      base[half * 8 + r] = wv8[r];
      base[16 + half * 8 + r] = cv8[r];
    }
  }
  float wv = base[lq];
  float cv = base[16 + lq];
#pragma unroll
  for (int j = 0; j < 32; ++j) acc[j] = acc[j] * cv + wv * t[j];
}

extern "C" __global__ void __launch_bounds__(256, 4)
hyp_attn_kernel(const float* __restrict__ cur, const float* __restrict__ hist,
                const float* __restrict__ curv,
                const float* __restrict__ Wq, const float* __restrict__ bq,
                const float* __restrict__ Wk, const float* __restrict__ bk,
                const float* __restrict__ av, float* __restrict__ out) {
  __shared__ float sw[8][64];  // [wave][parity*32 + {w:0..15, corr:16..31}]

  const int tid = threadIdx.x;
  const int wave = tid >> 5;
  const int lane = tid & 31;
  const int lq = lane & 15;
  const int half = lane >> 4;
  const int kb = half * 8;
  const int row = blockIdx.x * 128 + wave * 16 + lq;

  const float c = curv[0];
  const float sqrt_c = sqrtf(c);
  const float tos = 2.f / sqrt_c;

  float bkT[4], avT[4];
#pragma unroll
  for (int t4 = 0; t4 < 4; ++t4) {
    bkT[t4] = bk[t4 * 16 + lq];
    avT[t4] = av[t4 * 16 + lq];
  }

  // Resident weight fragments: Wk lives for the whole h-loop.
  v16h Bk_[4][2];
  load_wfrags(Wk, lq, kb, Bk_);

  // ---- query = log0(current) @ Wq^T + bq, produced directly in D-frag layout,
  // then packed to f16 (2 VGPR-octets instead of 4) for the resident copy.
  // Wq fragments are scoped here so their registers die before the main loop.
  v16h qh0, qh1;
  {
    v16h Bq_[4][2];
    load_wfrags(Wq, lq, kb, Bq_);
    float bqT[4];
#pragma unroll
    for (int t4 = 0; t4 < 4; ++t4) bqT[t4] = bq[t4 * 16 + lq];

    float tc[32];
    load_tile(tc, cur + (size_t)row * DD + kb);
    float ss = 0.f;
#pragma unroll
    for (int j = 0; j < 32; ++j) ss += tc[j] * tc[j];
    ss += __shfl_xor(ss, 16, 32);
    float nrm = sqrtf(ss);
    float scq = tos * atanhf(sqrt_c * nrm) / nrm;
#pragma unroll
    for (int j = 0; j < 32; ++j) tc[j] *= scq;

    v16h a0, a1;
#pragma unroll
    for (int j = 0; j < 16; ++j) { a0[j] = (_Float16)tc[j]; a1[j] = (_Float16)tc[16 + j]; }
#pragma unroll
    for (int t4 = 0; t4 < 4; ++t4) {
      v8f cb;
#pragma unroll
      for (int r = 0; r < 8; ++r) cb[r] = bqT[t4];
      cb = wmma_f16(a0, Bq_[t4][0], cb);
      cb = wmma_f16(a1, Bq_[t4][1], cb);
#pragma unroll
      for (int r = 0; r < 8; ++r) {
        if (t4 < 2) qh0[(t4 & 1) * 8 + r] = (_Float16)cb[r];
        else        qh1[(t4 & 1) * 8 + r] = (_Float16)cb[r];
      }
    }
  }

  // ---- streaming state
  float acc[32];
#pragma unroll
  for (int j = 0; j < 32; ++j) acc[j] = 0.f;
  float mrun[8], lrun[8];
#pragma unroll
  for (int r = 0; r < 8; ++r) { mrun[r] = -1e30f; lrun[r] = 0.f; }

  volatile float* swb = sw[wave];

  // ---- single pass over history: 1 tile buffer + depth-2 L2 prefetch.
  // Keep the loop ROLLED: full unroll merges tile live ranges and spills.
  const size_t hstep = (size_t)NN * DD;              // elements per h slice
  const float* hp0 = hist + (size_t)row * DD + kb;   // this lane's slice base

#pragma unroll 1
  for (int h = 0; h < HH; ++h) {
    {
      const char* pf =
          (const char*)(hp0 + ((h + 2 < HH) ? 2 * hstep : 0) - (size_t)kb);
      __builtin_prefetch(pf, 0, 0);
      __builtin_prefetch(pf + 128, 0, 0);
    }
    float t[32];
    load_tile(t, hp0);
    process_tile(t, h & 1, lq, half, Bk_, qh0, qh1, bkT, avT,
                 sqrt_c, tos, mrun, lrun, acc, swb);
    hp0 += hstep;
  }

  // ---- softmax normalization (broadcast lrun), then exp map at origin
  if (lq == 0) {
#pragma unroll
    for (int r = 0; r < 8; ++r) swb[half * 8 + r] = lrun[r];
  }
  float linv = 1.f / swb[lq];
#pragma unroll
  for (int j = 0; j < 32; ++j) acc[j] *= linv;

  float ss = 0.f;
#pragma unroll
  for (int j = 0; j < 32; ++j) ss += acc[j] * acc[j];
  ss += __shfl_xor(ss, 16, 32);
  float nrm = sqrtf(ss);
  float g = htanh(0.5f * sqrt_c * nrm) / (sqrt_c * nrm);
#pragma unroll
  for (int j = 0; j < 32; ++j) acc[j] *= g;

  float* orow = out + (size_t)row * DD + kb;
#pragma unroll
  for (int c2 = 0; c2 < 4; ++c2) {
    float4 x = make_float4(acc[c2 * 8 + 0], acc[c2 * 8 + 1], acc[c2 * 8 + 2], acc[c2 * 8 + 3]);
    float4 y = make_float4(acc[c2 * 8 + 4], acc[c2 * 8 + 5], acc[c2 * 8 + 6], acc[c2 * 8 + 7]);
    *(float4*)(orow + c2 * 16) = x;
    *(float4*)(orow + c2 * 16 + 4) = y;
  }
}

extern "C" void kernel_launch(void* const* d_in, const int* in_sizes, int n_in,
                              void* d_out, int out_size, void* d_ws, size_t ws_size,
                              hipStream_t stream) {
  (void)in_sizes; (void)n_in; (void)out_size; (void)d_ws; (void)ws_size;
  const float* cur  = (const float*)d_in[0];
  const float* hist = (const float*)d_in[1];
  const float* curv = (const float*)d_in[2];
  const float* Wq   = (const float*)d_in[3];
  const float* bq   = (const float*)d_in[4];
  const float* Wk   = (const float*)d_in[5];
  const float* bk   = (const float*)d_in[6];
  const float* av   = (const float*)d_in[7];
  float* out = (float*)d_out;

  dim3 grid(NN / 128), block(256);
  hyp_attn_kernel<<<grid, block, 0, stream>>>(cur, hist, curv, Wq, bq, Wk, bk, av, out);
}
